// NeuralODEBlock_89945205113172
// MI455X (gfx1250) — matrix operand order repaired and checked by reading, not compiled, not run on hardware
//
#include <hip/hip_runtime.h>
#include <math.h>

#define BATCH  8192
#define HID    512
#define H2     (2 * HID)
#define NSTEPS 10

typedef __attribute__((ext_vector_type(8)))  _Float16 v8h;
typedef __attribute__((ext_vector_type(16))) _Float16 v16h;
typedef __attribute__((ext_vector_type(8)))  float    v8f;

__device__ __forceinline__ v16h join16(v8h lo, v8h hi) {
  return __builtin_shufflevector(lo, hi, 0,1,2,3,4,5,6,7,8,9,10,11,12,13,14,15);
}

// -----------------------------------------------------------------------------
// NT GEMM, fused bias (+optional tanh): out[m][n] = f( sum_k A[m][k]*W[n][k] + b[n] )
// A: M x K row-major f16.  W: N x K row-major f16.
//
// Wave tile 32(M) x 64(N): 2 A frags x 4 B frags -> 8 WMMAs per 12 b128 loads.
// 8 waves arranged 4(M) x 2(N) -> block tile 128 x 128.
// Software-pipelined: iteration k+1's 12 fragment loads are issued BEFORE the
// 8 WMMAs of iteration k, so the s_wait before the WMMA block only drains the
// previous iteration's loads (partial loadcnt wait), not the in-flight ones.
//
// CDNA5 wave32 f16 fragment layouts (ISA 7.12.2):
//   A 16x32: lane(l<16) row=l, v[0:3]=K 0..7, v[4:7]=K 16..23; lane l+16: K 8..15/24..31
//     -> code uses contiguous K = (l>>4)*16 + 0..15 instead: 16B loads at +0 and +8.
//   B 32x16: lane holds col n=l&15, this code uses a contiguous K run of 16 at K=(l>>4)*16, matching A
//     -> two contiguous 16B loads.
//   C/D 16x16 f32: VGPR r -> (m = r + 8*(l>>4), n = l&15).
// -----------------------------------------------------------------------------
template<bool TANH, bool OUT_F16>
__global__ __launch_bounds__(256) void gemm_nt(
    const _Float16* __restrict__ A, const _Float16* __restrict__ W,
    const float* __restrict__ bias, _Float16* __restrict__ outH,
    float* __restrict__ outF, int M, int N, int K)
{
  const int lane   = threadIdx.x & 31;
  const int wave   = threadIdx.x >> 5;
  const int mBase  = blockIdx.x * 128 + (wave & 3) * 32;   // wave tile M origin
  const int nBase  = blockIdx.y * 128 + (wave >> 2) * 64;  // wave tile N origin
  const int laneLo = lane & 15;
  const int laneHi = lane >> 4;
  (void)M;

  const _Float16* aP = A + (size_t)(mBase + laneLo) * K + laneHi * 16;
  const _Float16* bP = W + (size_t)(nBase + laneLo) * K + laneHi * 16;
  const size_t tStride = (size_t)16 * K;   // one 16-row subtile in A or W

  auto ldA = [&](int mi, int k) {
    const _Float16* p = aP + (size_t)mi * tStride + k;
    return join16(*(const v8h*)p, *(const v8h*)(p + 8));
  };
  auto ldB = [&](int ni, int k) {
    const _Float16* p = bP + (size_t)ni * tStride + k;
    return join16(*(const v8h*)p, *(const v8h*)(p + 8));
  };

  v8f acc[2][4] = {};

  // prologue: fragments for k = 0
  v16h a0 = ldA(0, 0), a1 = ldA(1, 0);
  v16h b0 = ldB(0, 0), b1 = ldB(1, 0), b2 = ldB(2, 0), b3 = ldB(3, 0);

  for (int k = 32; k < K; k += 32) {
    // issue next iteration's loads first (stay in flight across the WMMAs)
    v16h a0n = ldA(0, k), a1n = ldA(1, k);
    v16h b0n = ldB(0, k), b1n = ldB(1, k), b2n = ldB(2, k), b3n = ldB(3, k);

    acc[0][0] = __builtin_amdgcn_wmma_f32_16x16x32_f16(false, a0, false, b0, (short)0, acc[0][0], false, false);
    acc[0][1] = __builtin_amdgcn_wmma_f32_16x16x32_f16(false, a0, false, b1, (short)0, acc[0][1], false, false);
    acc[0][2] = __builtin_amdgcn_wmma_f32_16x16x32_f16(false, a0, false, b2, (short)0, acc[0][2], false, false);
    acc[0][3] = __builtin_amdgcn_wmma_f32_16x16x32_f16(false, a0, false, b3, (short)0, acc[0][3], false, false);
    acc[1][0] = __builtin_amdgcn_wmma_f32_16x16x32_f16(false, a1, false, b0, (short)0, acc[1][0], false, false);
    acc[1][1] = __builtin_amdgcn_wmma_f32_16x16x32_f16(false, a1, false, b1, (short)0, acc[1][1], false, false);
    acc[1][2] = __builtin_amdgcn_wmma_f32_16x16x32_f16(false, a1, false, b2, (short)0, acc[1][2], false, false);
    acc[1][3] = __builtin_amdgcn_wmma_f32_16x16x32_f16(false, a1, false, b3, (short)0, acc[1][3], false, false);

    a0 = a0n; a1 = a1n; b0 = b0n; b1 = b1n; b2 = b2n; b3 = b3n;
  }

  // epilogue K-step
  acc[0][0] = __builtin_amdgcn_wmma_f32_16x16x32_f16(false, a0, false, b0, (short)0, acc[0][0], false, false);
  acc[0][1] = __builtin_amdgcn_wmma_f32_16x16x32_f16(false, a0, false, b1, (short)0, acc[0][1], false, false);
  acc[0][2] = __builtin_amdgcn_wmma_f32_16x16x32_f16(false, a0, false, b2, (short)0, acc[0][2], false, false);
  acc[0][3] = __builtin_amdgcn_wmma_f32_16x16x32_f16(false, a0, false, b3, (short)0, acc[0][3], false, false);
  acc[1][0] = __builtin_amdgcn_wmma_f32_16x16x32_f16(false, a1, false, b0, (short)0, acc[1][0], false, false);
  acc[1][1] = __builtin_amdgcn_wmma_f32_16x16x32_f16(false, a1, false, b1, (short)0, acc[1][1], false, false);
  acc[1][2] = __builtin_amdgcn_wmma_f32_16x16x32_f16(false, a1, false, b2, (short)0, acc[1][2], false, false);
  acc[1][3] = __builtin_amdgcn_wmma_f32_16x16x32_f16(false, a1, false, b3, (short)0, acc[1][3], false, false);

  // epilogue: bias + activation + store
#pragma unroll
  for (int mi = 0; mi < 2; ++mi) {
    const int mRow = mBase + mi * 16 + laneHi * 8;
#pragma unroll
    for (int ni = 0; ni < 4; ++ni) {
      const int n = nBase + ni * 16 + laneLo;
      const float bb = bias[n];
#pragma unroll
      for (int r = 0; r < 8; ++r) {
        float v = acc[mi][ni][r] + bb;
        if (TANH) v = tanhf(v);
        const size_t idx = (size_t)(mRow + r) * N + n;
        if (OUT_F16) outH[idx] = (_Float16)v;
        else         outF[idx] = v;
      }
    }
  }
}

// f32 -> f16 weight conversion (once per launch; weights then live in L2)
__global__ void cvt_f16_kernel(const float* __restrict__ in, _Float16* __restrict__ out, int n) {
  int i = blockIdx.x * blockDim.x + threadIdx.x;
  if (i < n) out[i] = (_Float16)in[i];
}

// Fused RK4 input prep: x = h + kx*k + t*Wt[col] + bt[col]  (cast to f16),
// and ksum = (init ? ks*k : ksum + ks*k) when useK.
__global__ void prep_kernel(const float* __restrict__ hIn, const float* __restrict__ kIn,
                            float* __restrict__ kSum, _Float16* __restrict__ xOut,
                            const float* __restrict__ Wt, const float* __restrict__ bt,
                            float t, float kx, float ks, int initSum, int useK, int total)
{
  int i = blockIdx.x * blockDim.x + threadIdx.x;
  if (i >= total) return;
  const int col = i % HID;
  float x = hIn[i] + t * Wt[col] + bt[col];
  if (useK) {
    const float kv = kIn[i];
    x += kx * kv;
    if (initSum) kSum[i] = ks * kv;
    else         kSum[i] += ks * kv;
  }
  xOut[i] = (_Float16)x;
}

// h_out = h + dt/6 * (k1 + 2k2 + 2k3 + k4) ; kSum holds k1+2k2+2k3, kLast holds k4
__global__ void finalize_kernel(const float* __restrict__ hIn, const float* __restrict__ kSum,
                                const float* __restrict__ kLast, float* __restrict__ hOut,
                                float dt6, int total)
{
  int i = blockIdx.x * blockDim.x + threadIdx.x;
  if (i < total) hOut[i] = hIn[i] + dt6 * (kSum[i] + kLast[i]);
}

extern "C" void kernel_launch(void* const* d_in, const int* in_sizes, int n_in,
                              void* d_out, int out_size, void* d_ws, size_t ws_size,
                              hipStream_t stream) {
  (void)in_sizes; (void)n_in; (void)out_size; (void)ws_size;

  const float* h  = (const float*)d_in[0];
  const float* W1 = (const float*)d_in[1];
  const float* b1 = (const float*)d_in[2];
  const float* W2 = (const float*)d_in[3];
  const float* b2 = (const float*)d_in[4];
  const float* W3 = (const float*)d_in[5];
  const float* b3 = (const float*)d_in[6];
  const float* Wt = (const float*)d_in[7];
  const float* bt = (const float*)d_in[8];
  float* out = (float*)d_out;

  // Workspace carve-out (256B aligned). Total ~92 MB.
  char* ws = (char*)d_ws;
  size_t off = 0;
  auto take = [&](size_t bytes) -> void* {
    void* p = ws + off;
    off += (bytes + 255) & ~(size_t)255;
    return p;
  };
  _Float16* W1h  = (_Float16*)take((size_t)H2 * HID * 2);
  _Float16* W2h  = (_Float16*)take((size_t)H2 * H2  * 2);
  _Float16* W3h  = (_Float16*)take((size_t)HID * H2 * 2);
  _Float16* x0   = (_Float16*)take((size_t)BATCH * HID * 2);
  _Float16* y1   = (_Float16*)take((size_t)BATCH * H2  * 2);
  _Float16* y2   = (_Float16*)take((size_t)BATCH * H2  * 2);
  float*    kbuf = (float*)take((size_t)BATCH * HID * 4);
  float*    ksum = (float*)take((size_t)BATCH * HID * 4);
  float*    hbuf = (float*)take((size_t)BATCH * HID * 4);

  const dim3 blk(256);

  // One-time (per launch) f32 -> f16 weight conversion
  {
    const int n1 = H2 * HID, n2 = H2 * H2, n3 = HID * H2;
    cvt_f16_kernel<<<(n1 + 255) / 256, blk, 0, stream>>>(W1, W1h, n1);
    cvt_f16_kernel<<<(n2 + 255) / 256, blk, 0, stream>>>(W2, W2h, n2);
    cvt_f16_kernel<<<(n3 + 255) / 256, blk, 0, stream>>>(W3, W3h, n3);
  }

  const float dt  = 1.0f / NSTEPS;
  const int   tot = BATCH * HID;
  const dim3  gE((tot + 255) / 256);
  const dim3  gG1(BATCH / 128, H2 / 128);   // GEMM1/2: N = 1024 -> (64, 8)
  const dim3  gG3(BATCH / 128, HID / 128);  // GEMM3:   N = 512  -> (64, 4)

  struct Ev { float t, kx, ks; int init, useK; };

  for (int s = 0; s < NSTEPS; ++s) {
    const float  t0  = s * dt;
    const float* hIn = (s == 0) ? h : hbuf;
    const Ev evs[4] = {
      { t0,             0.0f,      0.0f, 0, 0 },  // k1
      { t0 + 0.5f*dt,   0.5f*dt,   1.0f, 1, 1 },  // k2: x=h+dt/2*k1 ; ksum=k1
      { t0 + 0.5f*dt,   0.5f*dt,   2.0f, 0, 1 },  // k3: x=h+dt/2*k2 ; ksum+=2k2
      { t0 + dt,        dt,        2.0f, 0, 1 },  // k4: x=h+dt*k3   ; ksum+=2k3
    };

    for (int e = 0; e < 4; ++e) {
      prep_kernel<<<gE, blk, 0, stream>>>(hIn, kbuf, ksum, x0, Wt, bt,
                                          evs[e].t, evs[e].kx, evs[e].ks,
                                          evs[e].init, evs[e].useK, tot);
      gemm_nt<true,  true ><<<gG1, blk, 0, stream>>>(x0, W1h, b1, y1, nullptr, BATCH, H2,  HID);
      gemm_nt<true,  true ><<<gG1, blk, 0, stream>>>(y1, W2h, b2, y2, nullptr, BATCH, H2,  H2);
      gemm_nt<false, false><<<gG3, blk, 0, stream>>>(y2, W3h, b3, nullptr, kbuf, BATCH, HID, H2);
    }

    float* hOut = (s == NSTEPS - 1) ? out : hbuf;
    finalize_kernel<<<gE, blk, 0, stream>>>(hIn, ksum, kbuf, hOut, dt / 6.0f, tot);
  }
}
